// SCCA_2826088480860
// MI455X (gfx1250) — matrix operand order repaired and checked by reading, not compiled, not run on hardware
//
#include <hip/hip_runtime.h>
#include <math.h>

// Problem sizes (match reference)
#define Bsz 8
#define Nn  8192
#define Cc  256
#define Hh  8
#define TT  64          // tokens per workgroup
#define THREADS 256     // 8 waves (wave32)

typedef __attribute__((ext_vector_type(16))) _Float16 v16h;
typedef __attribute__((ext_vector_type(8)))  _Float16 v8h;
typedef __attribute__((ext_vector_type(8)))  float    v8f;

typedef __attribute__((address_space(3))) _Float16 lds_f16;

// ---- swizzled-weight workspace layout (in halves) ----
#define SZ_FC1  (512*256)
#define SZ_W    (256*256)
#define OFF_FC1 0
#define OFF_FC2 (OFF_FC1+SZ_FC1)
#define OFF_Q01 (OFF_FC2+SZ_W)
#define OFF_K01 (OFF_Q01+SZ_W)
#define OFF_V01 (OFF_K01+SZ_W)
#define OFF_O01 (OFF_V01+SZ_W)
#define OFF_Q10 (OFF_O01+SZ_W)
#define OFF_K10 (OFF_Q10+SZ_W)
#define OFF_V10 (OFF_K10+SZ_W)
#define OFF_O10 (OFF_V10+SZ_W)

// ---- LDS layout (bytes) ----
#define L_X0   0
#define L_X1   32768
#define L_H    65536     // h, then xr = x*rel
#define L_L    98304     // logits f16, then ctx f16
#define L_Q    131072
#define L_K0   163840
#define L_K1   196608
#define L_AB   229376    // attention probs: TT*H*2 floats = 4096 B
#define L_KN   233472    // kn f16: 2*256 halves = 1024 B
#define L_VN   234496    // vn f16: 2*256 halves = 1024 B
#define L_BS   235520    // B-stage double buffer: 2 x 16384 B
#define SMEM_BYTES (235520 + 32768)

// -----------------------------------------------------------------------------
// Weight prep: f32 (out,in) row-major, B = W^T (K x 256), swizzled into the
// WMMA 16-bit B-fragment order: block (kt,nt) of 512 halves; within block,
// half index = lane*16 + i  with  K = kt*32 + (i<8 ? (lane/16)*8 + i : 16 + (lane/16)*8 + i-8), N = nt*16 + lane%16
// A K-slice (fixed kt, all 16 nt) is a contiguous 16 KB run -> async-stageable.
// -----------------------------------------------------------------------------
__global__ void prep_w(const float* __restrict__ src, _Float16* __restrict__ dst,
                       int rowOff, int srcLd, int totalHalves) {
    int idx = blockIdx.x * 256 + threadIdx.x;
    if (idx >= totalHalves) return;
    int i    = idx & 15;
    int lane = (idx >> 4) & 31;
    int blk  = idx >> 9;
    int nt   = blk & 15;
    int kt   = blk >> 4;
    int k = kt * 32 + (lane >> 4) * 8 + (i & 7) + (i >> 3) * 16;
    int n = nt * 16 + (lane & 15);
    dst[idx] = (_Float16)src[(size_t)(rowOff + n) * srcLd + k];
}

// -----------------------------------------------------------------------------
// A-fragment load from a row-major f16 LDS tile (stride 256 halves).
// 16-bit A 16x32 layout: lane L: M = L%16; halves 0..7 -> K = g*8 + i,
// halves 8..15 -> K = 16 + g*8 + (i-8), all + ks*32.   (g = L/16)
// Two ds_load_b128 per fragment.
// -----------------------------------------------------------------------------
__device__ __forceinline__ v16h ldsA(const _Float16* tile, int mrow, int ks, int g) {
    const _Float16* p = tile + mrow * Cc + ks * 32 + g * 8;
    v8h lo = *(const v8h*)p;
    v8h hi = *(const v8h*)(p + 16);
    v16h r;
#pragma unroll
    for (int i = 0; i < 8; ++i) { r[i] = lo[i]; r[i + 8] = hi[i]; }
    return r;
}

__device__ __forceinline__ float gelu_exact(float x) {
    return 0.5f * x * (1.0f + erff(x * 0.70710678118654752f));
}

// -----------------------------------------------------------------------------
// Async copy of one 16 KB weight K-slice (8192 halves) into LDS.
// 1024 x 16B chunks, 4 per thread. Tracked by ASYNCcnt.
// -----------------------------------------------------------------------------
__device__ __forceinline__ void stageB(const _Float16* __restrict__ Wg, int kt,
                                       _Float16* dstLds, int tid) {
    const _Float16* src = Wg + (size_t)kt * 8192;
#pragma unroll
    for (int t = 0; t < 4; ++t) {
        int c = tid + t * 256;
        lds_f16* d = (lds_f16*)(dstLds + c * 8);
        const _Float16* s = src + c * 8;
        asm volatile("global_load_async_to_lds_b128 %0, %1, off"
                     :: "v"(d), "v"(s)
                     : "memory");
    }
}

__device__ __forceinline__ void wait_async0() {
    asm volatile("s_wait_asynccnt 0" ::: "memory");
}

#define WMMA(a, b, c) __builtin_amdgcn_wmma_f32_16x16x32_f16(false, (a), false, (b), (short)0, (c), false, false)

// -----------------------------------------------------------------------------
// Cooperative double-buffered GEMM: all 8 waves; B K-slices staged via async
// loads to LDS; each wave owns 8 N-tiles (ntBase) and one 16-token M-tile.
// -----------------------------------------------------------------------------
template<int KT, class AF, class EP>
__device__ __forceinline__ void gemmS(const _Float16* __restrict__ Wg,
                                      _Float16* bst, int tid, int lane, int ntBase,
                                      AF&& af, EP&& ep) {
    stageB(Wg, 0, bst, tid);
    wait_async0();
    __syncthreads();
    v8f acc[8] = {};
    for (int kt = 0; kt < KT; ++kt) {
        _Float16* bb = bst + (kt & 1) * 8192;
        if (kt + 1 < KT) stageB(Wg, kt + 1, bst + ((kt + 1) & 1) * 8192, tid);
        v16h a = af(kt);
#pragma unroll
        for (int j = 0; j < 8; ++j) {
            v16h b = ((const v16h*)bb)[(size_t)(ntBase + j) * 32 + lane];
            acc[j] = WMMA(a, b, acc[j]);
        }
        wait_async0();
        __syncthreads();
    }
#pragma unroll
    for (int j = 0; j < 8; ++j) ep(ntBase + j, acc[j]);
}

// Dual-A variant (shared B fragment feeds two accumulators) for fused V0/V1.
template<class AF0, class AF1, class EP>
__device__ __forceinline__ void gemmS2(const _Float16* __restrict__ Wg,
                                       _Float16* bst, int tid, int lane, int ntBase,
                                       AF0&& af0, AF1&& af1, EP&& ep) {
    stageB(Wg, 0, bst, tid);
    wait_async0();
    __syncthreads();
    v8f c0[8] = {}, c1[8] = {};
    for (int kt = 0; kt < 8; ++kt) {
        _Float16* bb = bst + (kt & 1) * 8192;
        if (kt + 1 < 8) stageB(Wg, kt + 1, bst + ((kt + 1) & 1) * 8192, tid);
        v16h a0 = af0(kt);
        v16h a1 = af1(kt);
#pragma unroll
        for (int j = 0; j < 8; ++j) {
            v16h b = ((const v16h*)bb)[(size_t)(ntBase + j) * 32 + lane];
            c0[j] = WMMA(a0, b, c0[j]);
            c1[j] = WMMA(a1, b, c1[j]);
        }
        wait_async0();
        __syncthreads();
    }
#pragma unroll
    for (int j = 0; j < 8; ++j) ep(ntBase + j, c0[j], c1[j]);
}

// -----------------------------------------------------------------------------
// Fused SCCA kernel: one workgroup = 64 tokens, both directions.
// wave w: mt = w>>1 (16-token M tile), ntBase = (w&1)*8 (8 of 16 N tiles)
// -----------------------------------------------------------------------------
__global__ __launch_bounds__(THREADS, 1)
void scca_fused(const float* __restrict__ x0, const float* __restrict__ x1,
                const float* __restrict__ fc1_b, const float* __restrict__ fc2_b,
                const float* __restrict__ inb01, const float* __restrict__ outb01,
                const float* __restrict__ inb10, const float* __restrict__ outb10,
                const float* __restrict__ kno, const float* __restrict__ vno,
                const _Float16* __restrict__ wsw, float* __restrict__ y) {
    extern __shared__ char smem[];
    _Float16* x0h  = (_Float16*)(smem + L_X0);
    _Float16* x1h  = (_Float16*)(smem + L_X1);
    _Float16* hbuf = (_Float16*)(smem + L_H);
    _Float16* lbuf = (_Float16*)(smem + L_L);
    _Float16* Qb   = (_Float16*)(smem + L_Q);
    _Float16* K0b  = (_Float16*)(smem + L_K0);
    _Float16* K1b  = (_Float16*)(smem + L_K1);
    float*    abuf = (float*)   (smem + L_AB);
    _Float16* knh  = (_Float16*)(smem + L_KN);
    _Float16* vnh  = (_Float16*)(smem + L_VN);
    _Float16* bst  = (_Float16*)(smem + L_BS);

    const int tid    = threadIdx.x;
    const int lane   = tid & 31;
    const int wave   = tid >> 5;
    const int mt     = wave >> 1;
    const int ntBase = (wave & 1) * 8;
    const int g      = lane >> 4;
    const int ln     = lane & 15;
    const size_t tok0 = (size_t)blockIdx.x * TT;

    // keep hot weight regions resident (global_prefetch_b8)
    __builtin_prefetch(wsw + OFF_FC1, 0, 3);
    __builtin_prefetch(wsw + OFF_Q01, 0, 3);
    __builtin_prefetch(wsw + OFF_Q10, 0, 3);

    // ---- stage x tiles f32 -> f16 LDS; stage noise vectors ----
    {
        const float4* g0 = (const float4*)(x0 + tok0 * Cc);
        const float4* g1 = (const float4*)(x1 + tok0 * Cc);
        for (int idx = tid; idx < TT * Cc / 4; idx += THREADS) {
            float4 a = g0[idx];
            float4 b = g1[idx];
            _Float16* p0 = x0h + idx * 4;
            p0[0] = (_Float16)a.x; p0[1] = (_Float16)a.y;
            p0[2] = (_Float16)a.z; p0[3] = (_Float16)a.w;
            _Float16* p1 = x1h + idx * 4;
            p1[0] = (_Float16)b.x; p1[1] = (_Float16)b.y;
            p1[2] = (_Float16)b.z; p1[3] = (_Float16)b.w;
        }
        knh[tid] = (_Float16)kno[tid];  knh[tid + 256] = (_Float16)kno[tid + 256];
        vnh[tid] = (_Float16)vno[tid];  vnh[tid + 256] = (_Float16)vno[tid + 256];
    }
    __syncthreads();

    for (int d = 0; d < 2; ++d) {
        const _Float16* xa  = d ? x1h : x0h;   // "self" stream
        const _Float16* xb  = d ? x0h : x1h;   // "other" stream
        const _Float16* Wq  = wsw + (d ? OFF_Q10 : OFF_Q01);
        const _Float16* Wk  = wsw + (d ? OFF_K10 : OFF_K01);
        const _Float16* Wv  = wsw + (d ? OFF_V10 : OFF_V01);
        const _Float16* Wo  = wsw + (d ? OFF_O10 : OFF_O01);
        const float*    inb  = d ? inb10 : inb01;
        const float*    outb = d ? outb10 : outb01;
        const _Float16* kn  = knh + d * Cc;
        const _Float16* vn  = vnh + d * Cc;
        const float*    xg  = d ? x1 : x0;
        float* yout = y + (size_t)d * Bsz * Nn * Cc;
        const _Float16* Aa = xa + mt * 16 * Cc;
        const _Float16* Ab = xb + mt * 16 * Cc;

        // ---- judge GEMM1: h = gelu(cat(xa,xb) @ fc1^T + fc1_b), K = 512 ----
        gemmS<16>(wsw + OFF_FC1, bst, tid, lane, ntBase,
            [&](int kt) { return kt < 8 ? ldsA(Aa, ln, kt, g) : ldsA(Ab, ln, kt - 8, g); },
            [&](int nt, v8f acc) {
                int col = nt * 16 + ln;
                float bias = fc1_b[col];
#pragma unroll
                for (int v = 0; v < 8; ++v) {
                    int row = mt * 16 + v + 8 * g;
                    hbuf[row * Cc + col] = (_Float16)gelu_exact(acc[v] + bias);
                }
            });

        // ---- judge GEMM2: logits = h @ fc2^T + fc2_b ----
        gemmS<8>(wsw + OFF_FC2, bst, tid, lane, ntBase,
            [&](int kt) { return ldsA(hbuf + mt * 16 * Cc, ln, kt, g); },
            [&](int nt, v8f acc) {
                int col = nt * 16 + ln;
                float bias = fc2_b[col];
#pragma unroll
                for (int v = 0; v < 8; ++v) {
                    int row = mt * 16 + v + 8 * g;
                    lbuf[row * Cc + col] = (_Float16)(acc[v] + bias);
                }
            });
        __syncthreads();

        // ---- channel softmax -> rel; write xr = xa * rel into hbuf ----
        if (tid < TT) {
            const _Float16* lr = lbuf + tid * Cc;
            float mx = -1e30f;
            for (int c = 0; c < Cc; ++c) mx = fmaxf(mx, (float)lr[c]);
            float s = 0.f;
            for (int c = 0; c < Cc; ++c) s += expf((float)lr[c] - mx);
            float inv = 1.f / s;
            const _Float16* xs = xa + tid * Cc;
            for (int c = 0; c < Cc; ++c) {
                float p = expf((float)lr[c] - mx) * inv;
                hbuf[tid * Cc + c] = (_Float16)((float)xs[c] * p);
            }
        }
        __syncthreads();

        // ---- Q = xa@Wq + bq ----
        gemmS<8>(Wq, bst, tid, lane, ntBase,
            [&](int kt) { return ldsA(Aa, ln, kt, g); },
            [&](int nt, v8f acc) {
                int col = nt * 16 + ln;
                float bias = inb[col];
#pragma unroll
                for (int v = 0; v < 8; ++v) {
                    int row = mt * 16 + v + 8 * g;
                    Qb[row * Cc + col] = (_Float16)(acc[v] + bias);
                }
            });

        // ---- K0 = (xa+kn)@Wk + bk  (packed f16 noise add on A fragments) ----
        gemmS<8>(Wk, bst, tid, lane, ntBase,
            [&](int kt) { return ldsA(Aa, ln, kt, g) + ldsA(kn, 0, kt, g); },
            [&](int nt, v8f acc) {
                int col = nt * 16 + ln;
                float bias = inb[256 + col];
#pragma unroll
                for (int v = 0; v < 8; ++v) {
                    int row = mt * 16 + v + 8 * g;
                    K0b[row * Cc + col] = (_Float16)(acc[v] + bias);
                }
            });

        // ---- K1 = xr@Wk + bk ----
        gemmS<8>(Wk, bst, tid, lane, ntBase,
            [&](int kt) { return ldsA(hbuf + mt * 16 * Cc, ln, kt, g); },
            [&](int nt, v8f acc) {
                int col = nt * 16 + ln;
                float bias = inb[256 + col];
#pragma unroll
                for (int v = 0; v < 8; ++v) {
                    int row = mt * 16 + v + 8 * g;
                    K1b[row * Cc + col] = (_Float16)(acc[v] + bias);
                }
            });
        __syncthreads();

        // ---- 2-slot attention probs per (token, head) ----
        for (int p = tid; p < TT * Hh; p += THREADS) {
            int t = p >> 3, hh = p & 7;
            const _Float16* q  = Qb  + t * Cc + hh * 32;
            const _Float16* k0 = K0b + t * Cc + hh * 32;
            const _Float16* k1 = K1b + t * Cc + hh * 32;
            float s0 = 0.f, s1 = 0.f;
#pragma unroll
            for (int dd = 0; dd < 32; ++dd) {
                float qf = (float)q[dd];
                s0 += qf * (float)k0[dd];
                s1 += qf * (float)k1[dd];
            }
            s0 *= 0.17677669529663688f;  // 1/sqrt(32)
            s1 *= 0.17677669529663688f;
            float m  = fmaxf(s0, s1);
            float e0 = expf(s0 - m), e1 = expf(s1 - m);
            float z  = 1.f / (e0 + e1);
            abuf[p * 2]     = e0 * z;
            abuf[p * 2 + 1] = e1 * z;
        }
        __syncthreads();

        // ---- fused V0/V1 GEMMs (shared B fragment) + ctx -> lbuf ----
        gemmS2(Wv, bst, tid, lane, ntBase,
            [&](int kt) { return ldsA(Aa, ln, kt, g) + ldsA(vn, 0, kt, g); },
            [&](int kt) { return ldsA(Ab, ln, kt, g); },
            [&](int nt, v8f c0, v8f c1) {
                int col = nt * 16 + ln;
                float bias = inb[512 + col];
                int head = nt >> 1;
#pragma unroll
                for (int v = 0; v < 8; ++v) {
                    int row = mt * 16 + v + 8 * g;
                    float p0 = abuf[(row * Hh + head) * 2];
                    float p1 = abuf[(row * Hh + head) * 2 + 1];
                    lbuf[row * Cc + col] = (_Float16)(p0 * (c0[v] + bias) + p1 * (c1[v] + bias));
                }
            });
        __syncthreads();

        // ---- out = ctx@Wo + bo + residual -> global ----
        gemmS<8>(Wo, bst, tid, lane, ntBase,
            [&](int kt) { return ldsA(lbuf + mt * 16 * Cc, ln, kt, g); },
            [&](int nt, v8f acc) {
                int col = nt * 16 + ln;
                float bias = outb[col];
#pragma unroll
                for (int v = 0; v < 8; ++v) {
                    int row = mt * 16 + v + 8 * g;
                    size_t gi = (tok0 + row) * Cc + col;
                    yout[gi] = acc[v] + bias + xg[gi];
                }
            });
        __syncthreads();  // before buffers are reused by the other direction
    }
}

// -----------------------------------------------------------------------------
extern "C" void kernel_launch(void* const* d_in, const int* in_sizes, int n_in,
                              void* d_out, int out_size, void* d_ws, size_t ws_size,
                              hipStream_t stream) {
    (void)in_sizes; (void)n_in; (void)out_size; (void)ws_size;
    const float* x0      = (const float*)d_in[0];
    const float* x1      = (const float*)d_in[1];
    const float* fc1_w   = (const float*)d_in[2];
    const float* fc1_b   = (const float*)d_in[3];
    const float* fc2_w   = (const float*)d_in[4];
    const float* fc2_b   = (const float*)d_in[5];
    const float* in_w_01 = (const float*)d_in[6];
    const float* in_b_01 = (const float*)d_in[7];
    const float* out_w_01= (const float*)d_in[8];
    const float* out_b_01= (const float*)d_in[9];
    const float* in_w_10 = (const float*)d_in[10];
    const float* in_b_10 = (const float*)d_in[11];
    const float* out_w_10= (const float*)d_in[12];
    const float* out_b_10= (const float*)d_in[13];
    const float* k_noise = (const float*)d_in[14];
    const float* v_noise = (const float*)d_in[15];
    float* y = (float*)d_out;
    _Float16* wp = (_Float16*)d_ws;

    const int T = 256;
    // weight swizzle (B = W^T, WMMA B-fragment order)
    prep_w<<<(SZ_FC1 + T - 1) / T, T, 0, stream>>>(fc1_w,    wp + OFF_FC1, 0,   512, SZ_FC1);
    prep_w<<<(SZ_W   + T - 1) / T, T, 0, stream>>>(fc2_w,    wp + OFF_FC2, 0,   256, SZ_W);
    prep_w<<<(SZ_W   + T - 1) / T, T, 0, stream>>>(in_w_01,  wp + OFF_Q01, 0,   256, SZ_W);
    prep_w<<<(SZ_W   + T - 1) / T, T, 0, stream>>>(in_w_01,  wp + OFF_K01, 256, 256, SZ_W);
    prep_w<<<(SZ_W   + T - 1) / T, T, 0, stream>>>(in_w_01,  wp + OFF_V01, 512, 256, SZ_W);
    prep_w<<<(SZ_W   + T - 1) / T, T, 0, stream>>>(out_w_01, wp + OFF_O01, 0,   256, SZ_W);
    prep_w<<<(SZ_W   + T - 1) / T, T, 0, stream>>>(in_w_10,  wp + OFF_Q10, 0,   256, SZ_W);
    prep_w<<<(SZ_W   + T - 1) / T, T, 0, stream>>>(in_w_10,  wp + OFF_K10, 256, 256, SZ_W);
    prep_w<<<(SZ_W   + T - 1) / T, T, 0, stream>>>(in_w_10,  wp + OFF_V10, 512, 256, SZ_W);
    prep_w<<<(SZ_W   + T - 1) / T, T, 0, stream>>>(out_w_10, wp + OFF_O10, 0,   256, SZ_W);

    const int nBlocks = (Bsz * Nn) / TT;  // 1024
    scca_fused<<<nBlocks, THREADS, SMEM_BYTES, stream>>>(
        x0, x1, fc1_b, fc2_b, in_b_01, out_b_01, in_b_10, out_b_10,
        k_noise, v_noise, wp, y);
}
